// BPR_76665166234050
// MI455X (gfx1250) — matrix operand order repaired and checked by reading, not compiled, not run on hardware
//
#include <hip/hip_runtime.h>
#include <hip/hip_bf16.h>
#include <math.h>

typedef __attribute__((ext_vector_type(16))) _Float16 v16h;
typedef __attribute__((ext_vector_type(8)))  float    v8f;

#define DIMS 64

// ---------------- generic element-wise kernels ----------------

__global__ void fill_zero_k(float* __restrict__ p, long long n) {
  long long i = (long long)blockIdx.x * blockDim.x + threadIdx.x;
  long long stride = (long long)gridDim.x * blockDim.x;
  for (; i < n; i += stride) p[i] = 0.0f;
}

__global__ void copy_k(float* __restrict__ dst, const float* __restrict__ src, long long n) {
  long long i = (long long)blockIdx.x * blockDim.x + threadIdx.x;
  long long stride = (long long)gridDim.x * blockDim.x;
  for (; i < n; i += stride) dst[i] = src[i];
}

__global__ void axpy_k(float* __restrict__ y, const float* __restrict__ x, float a, long long n) {
  long long i = (long long)blockIdx.x * blockDim.x + threadIdx.x;
  long long stride = (long long)gridDim.x * blockDim.x;
  for (; i < n; i += stride) y[i] += a * x[i];
}

// ---------------- SpMM: y[seg[e]] += val[e] * x[gat[e]] ----------------
// 16 threads per edge, float4 per thread. Index loads are identical across
// the 16 lanes of a half-wave -> broadcast, negligible extra traffic.
// Memory-bound: ~1.7GB per hop, atomics resolved in L2 (global_atomic_add_f32).

__global__ __launch_bounds__(256) void spmm_scatter_k(
    const int* __restrict__ seg, const int* __restrict__ gat,
    const float* __restrict__ val, const float* __restrict__ x,
    float* __restrict__ y, int E) {
  long long tid = (long long)blockIdx.x * blockDim.x + threadIdx.x;
  long long total = (long long)E * 16;
  if (tid >= total) return;
  int e = (int)(tid >> 4);
  int c = (int)(tid & 15);
  int s = seg[e];
  int g = gat[e];
  float v = val[e];
  const float4 m = ((const float4*)(x + (size_t)g * DIMS))[c];
  float* dst = y + (size_t)s * DIMS + c * 4;
  atomicAdd(dst + 0, v * m.x);
  atomicAdd(dst + 1, v * m.y);
  atomicAdd(dst + 2, v * m.z);
  atomicAdd(dst + 3, v * m.w);
}

// ---------------- BPR loss via WMMA ----------------
// One wave (32 threads) handles a tile of 16 batch rows.
// A = u tile (16x64 f16), B = ii / ij tiles (64x16 f16).
// Two chained v_wmma_f32_16x16x32_f16 give the 16x16 product; the diagonal
// is the 16 dot products. Row norms (reg term) computed in f32 from the same
// loaded values + shfl_xor(16) pair-combine.

__device__ inline float pick8(const v8f& c, int idx) {
  float r = c[0];
#pragma unroll
  for (int k = 1; k < 8; ++k) r = (idx == k) ? c[k] : r;
  return r;
}

__global__ __launch_bounds__(32) void bpr_wmma_k(
    const int* __restrict__ user, const int* __restrict__ itemi, const int* __restrict__ itemj,
    const float* __restrict__ gcnU, const float* __restrict__ gcnI,
    float* __restrict__ acc /* [0]=bpr_sum, [1]=reg_sum */) {
  const int lane = threadIdx.x;          // 0..31
  const int r    = lane & 15;            // tile row / col handled by this lane
  const bool hi  = lane >= 16;
  const int b0   = blockIdx.x * 16;

  const int uid = user[b0 + r];
  const int iid = itemi[b0 + r];
  const int jid = itemj[b0 + r];
  const float* __restrict__ urow = gcnU + (size_t)uid * DIMS;
  const float* __restrict__ irow = gcnI + (size_t)iid * DIMS;
  const float* __restrict__ jrow = gcnI + (size_t)jid * DIMS;

  float squ = 0.f, sqi = 0.f, sqj = 0.f;
  v16h a0, a1, bi0, bi1, bj0, bj1;

  // A order chosen here, not AMD's table order (16-bit A 16x32): lanes 0-15 hold K 0..15,
  // lanes 16-31 hold K 16..31. Fragment 1 is the same +32.
  const int ak = hi ? 16 : 0;
#pragma unroll
  for (int t = 0; t < 8; ++t) { float f = urow[ak + t];      squ += f * f; a0[t]     = (_Float16)f; }
#pragma unroll
  for (int t = 0; t < 8; ++t) { float f = urow[ak + 8 + t]; squ += f * f; a0[8 + t] = (_Float16)f; }
#pragma unroll
  for (int t = 0; t < 8; ++t) { float f = urow[32 + ak + t];      squ += f * f; a1[t]     = (_Float16)f; }
#pragma unroll
  for (int t = 0; t < 8; ++t) { float f = urow[32 + ak + 8 + t]; squ += f * f; a1[8 + t] = (_Float16)f; }

  // B order chosen here, same as A (32x16): lanes 0-15 hold K=0..15 of column r, lanes 16-31 K=16..31.
  const int bk = hi ? 16 : 0;
#pragma unroll
  for (int t = 0; t < 16; ++t) { float f = irow[bk + t];      sqi += f * f; bi0[t] = (_Float16)f; }
#pragma unroll
  for (int t = 0; t < 16; ++t) { float f = irow[32 + bk + t]; sqi += f * f; bi1[t] = (_Float16)f; }
#pragma unroll
  for (int t = 0; t < 16; ++t) { float f = jrow[bk + t];      sqj += f * f; bj0[t] = (_Float16)f; }
#pragma unroll
  for (int t = 0; t < 16; ++t) { float f = jrow[32 + bk + t]; sqj += f * f; bj1[t] = (_Float16)f; }

  // D = A*B (+C), K accumulated 0..31 then 32..63. EXEC is all-1s here (no divergence yet).
  v8f ci = {};
  ci = __builtin_amdgcn_wmma_f32_16x16x32_f16(false, a0, false, bi0, (short)0, ci, false, false);
  ci = __builtin_amdgcn_wmma_f32_16x16x32_f16(false, a1, false, bi1, (short)0, ci, false, false);
  v8f cj = {};
  cj = __builtin_amdgcn_wmma_f32_16x16x32_f16(false, a0, false, bj0, (short)0, cj, false, false);
  cj = __builtin_amdgcn_wmma_f32_16x16x32_f16(false, a1, false, bj1, (short)0, cj, false, false);

  // Combine the two half-row partial norms (lane p and lane p+16 cover disjoint K).
  squ += __shfl_xor(squ, 16, 32);
  sqi += __shfl_xor(sqi, 16, 32);
  sqj += __shfl_xor(sqj, 16, 32);

  // Diagonal of the 16x16 result: M=m,N=m lives at (VGPR m, lane m) for m<8 and
  // (VGPR m-8, lane m+16) for m>=8  => active lanes are 0..7 and 24..31.
  int idx = -1;
  if (lane < 8)        idx = lane;       // row = lane
  else if (lane >= 24) idx = lane - 24;  // row = lane - 16
  float bv = 0.f, rv = 0.f;
  if (idx >= 0) {
    float pi = pick8(ci, idx);
    float pj = pick8(cj, idx);
    float x  = pi - pj;
    // -log(sigmoid(x)) = softplus(-x), numerically stable form
    bv = fmaxf(-x, 0.f) + log1pf(expf(-fabsf(x)));
    rv = squ + sqi + sqj;  // this lane holds norms for its own row (r or r, via xor16)
  }
#pragma unroll
  for (int o = 16; o > 0; o >>= 1) {
    bv += __shfl_down(bv, o, 32);
    rv += __shfl_down(rv, o, 32);
  }
  if (lane == 0) {
    atomicAdd(&acc[0], bv);
    atomicAdd(&acc[1], rv);
  }
}

// ---------------- self-distillation loss ----------------
// One wave per row: ||old - gcn||_2 * n[row] / rows, block-accumulated in LDS,
// one global atomic per block.

__global__ __launch_bounds__(256) void self_loss_k(
    const float* __restrict__ oldE, const float* __restrict__ gcn,
    const float* __restrict__ nv, float inv_rows, float* __restrict__ acc, int rows) {
  __shared__ float bsum;
  if (threadIdx.x == 0) bsum = 0.f;
  __syncthreads();
  int w    = (int)(((long long)blockIdx.x * blockDim.x + threadIdx.x) >> 5);
  int lane = threadIdx.x & 31;
  float contrib = 0.f;
  if (w < rows) {
    const float* a = oldE + (size_t)w * DIMS;
    const float* g = gcn  + (size_t)w * DIMS;
    float2 av = *(const float2*)(a + lane * 2);
    float2 gv = *(const float2*)(g + lane * 2);
    float dx = av.x - gv.x, dy = av.y - gv.y;
    float s = dx * dx + dy * dy;
#pragma unroll
    for (int o = 16; o > 0; o >>= 1) s += __shfl_down(s, o, 32);
    if (lane == 0) contrib = sqrtf(s) * nv[w] * inv_rows;
  }
  if (lane == 0 && contrib != 0.f) atomicAdd(&bsum, contrib);
  __syncthreads();
  if (threadIdx.x == 0) atomicAdd(acc, bsum);
}

__global__ void finalize_k(const float* __restrict__ acc, float* __restrict__ out, float invB) {
  if (threadIdx.x == 0) {
    float bpr = acc[0] * invB;
    float reg = 1e-4f * acc[1] * invB;
    out[0] = bpr + reg;
    out[1] = 100.0f * (acc[2] + acc[3]);
    out[2] = 1.0f;
    out[3] = 1.0f;
  }
}

// ---------------- launcher ----------------

extern "C" void kernel_launch(void* const* d_in, const int* in_sizes, int n_in,
                              void* d_out, int out_size, void* d_ws, size_t ws_size,
                              hipStream_t stream) {
  const int*   user     = (const int*)d_in[0];
  const int*   item_i   = (const int*)d_in[1];
  const int*   item_j   = (const int*)d_in[2];
  const int*   edge_u   = (const int*)d_in[3];
  const int*   edge_i   = (const int*)d_in[4];
  const float* edge_val = (const float*)d_in[5];
  const float* user_emb = (const float*)d_in[6];
  const float* item_emb = (const float*)d_in[7];
  const float* oldU     = (const float*)d_in[8];
  const float* oldI     = (const float*)d_in[9];
  const float* nU       = (const float*)d_in[10];
  const float* nI       = (const float*)d_in[11];

  const int B  = in_sizes[0];
  const int E  = in_sizes[3];
  const int UN = in_sizes[6] / DIMS;
  const int IN_ = in_sizes[7] / DIMS;
  const long long uE = (long long)UN * DIMS;
  const long long iE = (long long)IN_ * DIMS;

  float* ws   = (float*)d_ws;
  float* gcnU = ws;
  float* gcnI = gcnU + uE;
  float* hU0  = gcnI + iE;   // g1u, later g3u
  float* hU1  = hU0 + uE;    // g2u
  float* hI0  = hU1 + uE;    // g1i, later g3i
  float* hI1  = hI0 + iE;    // g2i
  float* acc  = hI1 + iE;    // 4 accumulators

  const int TPB = 256;
  const int gsU = 4096, gsI = 2048;                      // grid-stride grids
  const int spmmBlocks = (int)(((long long)E * 16 + TPB - 1) / TPB);

  // zero accumulators (must happen every launch)
  fill_zero_k<<<1, 32, 0, stream>>>(acc, 4);

  // gcn = emb (layer-0 term)
  copy_k<<<gsU, TPB, 0, stream>>>(gcnU, user_emb, uE);
  copy_k<<<gsI, TPB, 0, stream>>>(gcnI, item_emb, iE);

  // hop 1
  fill_zero_k<<<gsU, TPB, 0, stream>>>(hU0, uE);
  spmm_scatter_k<<<spmmBlocks, TPB, 0, stream>>>(edge_u, edge_i, edge_val, item_emb, hU0, E);
  fill_zero_k<<<gsI, TPB, 0, stream>>>(hI0, iE);
  spmm_scatter_k<<<spmmBlocks, TPB, 0, stream>>>(edge_i, edge_u, edge_val, user_emb, hI0, E);
  axpy_k<<<gsU, TPB, 0, stream>>>(gcnU, hU0, 0.5f, uE);
  axpy_k<<<gsI, TPB, 0, stream>>>(gcnI, hI0, 0.5f, iE);

  // hop 2 (g2u from g1i, g2i from g1u)
  fill_zero_k<<<gsU, TPB, 0, stream>>>(hU1, uE);
  spmm_scatter_k<<<spmmBlocks, TPB, 0, stream>>>(edge_u, edge_i, edge_val, hI0, hU1, E);
  fill_zero_k<<<gsI, TPB, 0, stream>>>(hI1, iE);
  spmm_scatter_k<<<spmmBlocks, TPB, 0, stream>>>(edge_i, edge_u, edge_val, hU0, hI1, E);
  axpy_k<<<gsU, TPB, 0, stream>>>(gcnU, hU1, 1.0f / 3.0f, uE);
  axpy_k<<<gsI, TPB, 0, stream>>>(gcnI, hI1, 1.0f / 3.0f, iE);

  // hop 3 (g3u from g2i, g3i from g2u) — reuse hU0/hI0
  fill_zero_k<<<gsU, TPB, 0, stream>>>(hU0, uE);
  spmm_scatter_k<<<spmmBlocks, TPB, 0, stream>>>(edge_u, edge_i, edge_val, hI1, hU0, E);
  fill_zero_k<<<gsI, TPB, 0, stream>>>(hI0, iE);
  spmm_scatter_k<<<spmmBlocks, TPB, 0, stream>>>(edge_i, edge_u, edge_val, hU1, hI0, E);
  axpy_k<<<gsU, TPB, 0, stream>>>(gcnU, hU0, 0.25f, uE);
  axpy_k<<<gsI, TPB, 0, stream>>>(gcnI, hI0, 0.25f, iE);

  // BPR loss (WMMA dot products), 16 rows per single-wave block
  bpr_wmma_k<<<B / 16, 32, 0, stream>>>(user, item_i, item_j, gcnU, gcnI, acc);

  // self-distillation terms
  self_loss_k<<<(UN * 32 + TPB - 1) / TPB, TPB, 0, stream>>>(oldU, gcnU, nU, 1.0f / (float)UN, &acc[2], UN);
  self_loss_k<<<(IN_ * 32 + TPB - 1) / TPB, TPB, 0, stream>>>(oldI, gcnI, nI, 1.0f / (float)IN_, &acc[3], IN_);

  finalize_k<<<1, 32, 0, stream>>>(acc, (float*)d_out, 1.0f / (float)B);
}
